// CVPGeometricProductLayer_40656160423985
// MI455X (gfx1250) — matrix operand order repaired and checked by reading, not compiled, not run on hardware
//
#include <hip/hip_runtime.h>
#include <cstdint>

// ---------------------------------------------------------------------------
// CDNA5 (gfx1250, wave32) fully fused CVP geometric-product layer.
// f16 WMMA (v_wmma_f32_16x16x32_f16) with f32 accumulation; all intermediates
// staged in LDS (258.25 KB / workgroup, within the 320 KB WGP budget).
// ---------------------------------------------------------------------------

typedef __attribute__((ext_vector_type(16))) _Float16 v16h;
typedef __attribute__((ext_vector_type(4)))  _Float16 v4h;
typedef __attribute__((ext_vector_type(8)))  float    v8f;

union F16x16 { v16h h; uint4 u[2]; };

// Cl(3,0) Cayley table, blade order {1,e1,e2,e3,e12,e13,e23,e123}
__device__ constexpr int GPK[8][8] = {
  {0,1,2,3,4,5,6,7},
  {1,0,4,5,2,3,7,6},
  {2,4,0,6,1,7,3,5},
  {3,5,6,0,7,1,2,4},
  {4,2,1,7,0,6,5,3},
  {5,3,7,1,6,0,4,2},
  {6,7,3,2,5,4,0,1},
  {7,6,5,4,3,2,1,0},
};
__device__ constexpr float GPS[8][8] = {
  { 1, 1, 1, 1, 1, 1, 1, 1},
  { 1, 1, 1, 1, 1, 1, 1, 1},
  { 1,-1, 1, 1,-1,-1, 1,-1},
  { 1,-1,-1, 1, 1,-1,-1, 1},
  { 1,-1, 1, 1,-1,-1, 1,-1},
  { 1,-1,-1, 1, 1,-1,-1, 1},
  { 1, 1,-1, 1,-1, 1,-1,-1},
  { 1, 1,-1, 1,-1, 1,-1,-1},
};

// ----- LDS layout (bytes) --------------------------------------------------
#define SMEM_SA     0         // gp / vv stage     [4rt][8k][16row][64ch] f16  64KB
#define SMEM_SB     65536     // g2 stage          same layout            f16  64KB
#define SMEM_V16    131072    // v tile            [64row][8k][64ch]      f16  64KB
#define SMEM_WL     196608    // W_left^T-friendly [64n][64m]             f16   8KB
#define SMEM_WR     204800
#define SMEM_WG     212992
#define SMEM_WW     221184
#define SMEM_WS2V   229376    // W_s2v             [64n][128m]            f16  16KB
#define SMEM_S2VIN  245760    // s2v A input       [64row][128k]          f16  16KB
#define SMEM_BL     262144    // biases / a_norm (f32, 64 each)
#define SMEM_BR     262400
#define SMEM_BG     262656
#define SMEM_BW     262912
#define SMEM_BS     263168
#define SMEM_AN     263424
#define SMEM_NP     263680    // norm partials [64row][2] f32
#define SMEM_MEAN   264192    // per-row mean  [64] f32
#define SMEM_TOTAL  264448

// A-fragment (16x32 f16, ISA 7.12.2 layout): lane m = lane&15,
// V0-3 hold K = hi*8..hi*8+7, V4-7 hold K = 16+hi*8..23+hi*8  (hi = lane>>4)
__device__ __forceinline__ v16h lds_frag_a(const _Float16* base, int lane,
                                           int strideH, int kbase) {
  const _Float16* p = base + (lane & 15) * strideH + kbase + ((lane >> 4) & 1) * 8;
  F16x16 t;
  t.u[0] = *reinterpret_cast<const uint4*>(p);
  t.u[1] = *reinterpret_cast<const uint4*>(p + 16);
  return t.h;
}

// B-fragment (32x16 f16): lane n = lane&15, V0-3 hold K = hi*8..hi*8+7,
// V4-7 hold K = 16+hi*8..23+hi*8 (same K order as A)
__device__ __forceinline__ v16h lds_frag_b(const _Float16* base, int lane,
                                           int strideH, int kbase) {
  const _Float16* p = base + (lane & 15) * strideH + kbase + ((lane >> 4) & 1) * 8;
  F16x16 t;
  t.u[0] = *reinterpret_cast<const uint4*>(p);
  t.u[1] = *reinterpret_cast<const uint4*>(p + 16);
  return t.h;
}

__device__ __forceinline__ v8f wmma_f16(v16h a, v16h b, v8f c) {
  return __builtin_amdgcn_wmma_f32_16x16x32_f16(false, a, false, b,
                                                (short)0, c, false, false);
}

__device__ __forceinline__ void load_w_f16(const float* __restrict__ g,
                                           _Float16* d, int tid, int n4) {
  for (int idx = tid; idx < n4; idx += 256) {
    float4 t = reinterpret_cast<const float4*>(g)[idx];
    v4h h = { (_Float16)t.x, (_Float16)t.y, (_Float16)t.z, (_Float16)t.w };
    *reinterpret_cast<v4h*>(d + idx * 4) = h;
  }
}

__global__ void __launch_bounds__(256, 1)
cvp_gp_fused(const float* __restrict__ gS,  const float* __restrict__ gV,
             const float* __restrict__ gWl, const float* __restrict__ gbl,
             const float* __restrict__ gWr, const float* __restrict__ gbr,
             const float* __restrict__ gWg, const float* __restrict__ gbg,
             const float* __restrict__ gAn, const float* __restrict__ gWw,
             const float* __restrict__ gbw, const float* __restrict__ gWs2v,
             const float* __restrict__ gbs, float* __restrict__ out, int Btot) {
  extern __shared__ char smem[];
  _Float16* sA    = reinterpret_cast<_Float16*>(smem + SMEM_SA);
  _Float16* sB    = reinterpret_cast<_Float16*>(smem + SMEM_SB);
  _Float16* v16   = reinterpret_cast<_Float16*>(smem + SMEM_V16);
  _Float16* wl    = reinterpret_cast<_Float16*>(smem + SMEM_WL);
  _Float16* wr    = reinterpret_cast<_Float16*>(smem + SMEM_WR);
  _Float16* wg    = reinterpret_cast<_Float16*>(smem + SMEM_WG);
  _Float16* ww    = reinterpret_cast<_Float16*>(smem + SMEM_WW);
  _Float16* ws2v  = reinterpret_cast<_Float16*>(smem + SMEM_WS2V);
  _Float16* s2vin = reinterpret_cast<_Float16*>(smem + SMEM_S2VIN);
  float* bL   = reinterpret_cast<float*>(smem + SMEM_BL);
  float* bR   = reinterpret_cast<float*>(smem + SMEM_BR);
  float* bG   = reinterpret_cast<float*>(smem + SMEM_BG);
  float* bW   = reinterpret_cast<float*>(smem + SMEM_BW);
  float* bS   = reinterpret_cast<float*>(smem + SMEM_BS);
  float* aN   = reinterpret_cast<float*>(smem + SMEM_AN);
  float* npart= reinterpret_cast<float*>(smem + SMEM_NP);
  float* smean= reinterpret_cast<float*>(smem + SMEM_MEAN);

  const int tid  = threadIdx.x;
  const int row0 = blockIdx.x * 64;
  const int lane = tid & 31;
  const int wave = tid >> 5;
  const int rt   = wave >> 1;   // 16-row tile owned by this wave
  const int nh   = wave & 1;    // output-channel half (ntiles 0-1 or 2-3)
  const int m15  = lane & 15;
  const int hi8  = ((lane >> 4) & 1) * 8;

  // ---------------- phase 0: cooperative loads -----------------------------
  load_w_f16(gWl, wl, tid, 1024);
  load_w_f16(gWr, wr, tid, 1024);
  load_w_f16(gWg, wg, tid, 1024);
  load_w_f16(gWw, ww, tid, 1024);
  load_w_f16(gWs2v, ws2v, tid, 2048);
  if (tid < 64) {
    bL[tid] = gbl[tid]; bR[tid] = gbr[tid]; bG[tid] = gbg[tid];
    bW[tid] = gbw[tid]; bS[tid] = gbs[tid]; aN[tid] = gAn[tid];
  }
  // s tile: passthrough to out[0..B*64) + f16 into s2vin cols 64..127
  {
    const float4* sp = reinterpret_cast<const float4*>(gS + (size_t)row0 * 64);
    float4* op = reinterpret_cast<float4*>(out + (size_t)row0 * 64);
    for (int idx = tid; idx < 1024; idx += 256) {
      float4 t = sp[idx];
      op[idx] = t;
      int f = idx * 4, row = f >> 6, ch = f & 63;
      v4h h = { (_Float16)t.x, (_Float16)t.y, (_Float16)t.z, (_Float16)t.w };
      *reinterpret_cast<v4h*>(s2vin + row * 128 + 64 + ch) = h;
    }
  }
  // v tile -> f16 [row][blade][ch]
  {
    const float4* vp = reinterpret_cast<const float4*>(gV + (size_t)row0 * 512);
    for (int idx = tid; idx < 8192; idx += 256) {
      float4 t = vp[idx];
      int f = idx * 4, row = f >> 9, rem = f & 511, ch = rem >> 3, k0 = rem & 7;
      v16[(row * 8 + k0 + 0) * 64 + ch] = (_Float16)t.x;
      v16[(row * 8 + k0 + 1) * 64 + ch] = (_Float16)t.y;
      v16[(row * 8 + k0 + 2) * 64 + ch] = (_Float16)t.z;
      v16[(row * 8 + k0 + 3) * 64 + ch] = (_Float16)t.w;
    }
  }
  __syncthreads();

  // ---------------- stage 1: vl/vr GEMMs + geometric product ---------------
  for (int t = 0; t < 2; ++t) {
    const int nt = nh * 2 + t, nbase = nt * 16;
    v16h Bl0 = lds_frag_b(wl + nbase * 64, lane, 64, 0);
    v16h Bl1 = lds_frag_b(wl + nbase * 64, lane, 64, 32);
    v16h Br0 = lds_frag_b(wr + nbase * 64, lane, 64, 0);
    v16h Br1 = lds_frag_b(wr + nbase * 64, lane, 64, 32);
    v8f vrj[8];
    #pragma unroll
    for (int j = 0; j < 8; ++j) {
      const _Float16* ab = v16 + (rt * 128 + j) * 64;   // row stride = 512 halves
      v8f c = {0.f,0.f,0.f,0.f,0.f,0.f,0.f,0.f};
      c = wmma_f16(lds_frag_a(ab, lane, 512, 0),  Br0, c);
      c = wmma_f16(lds_frag_a(ab, lane, 512, 32), Br1, c);
      vrj[j] = c;
    }
    { float bv = bR[nbase + m15];
      #pragma unroll
      for (int r = 0; r < 8; ++r) vrj[0][r] += bv; }
    float gp[8][8] = {};
    #pragma unroll
    for (int i = 0; i < 8; ++i) {
      const _Float16* ab = v16 + (rt * 128 + i) * 64;
      v8f c = {0.f,0.f,0.f,0.f,0.f,0.f,0.f,0.f};
      c = wmma_f16(lds_frag_a(ab, lane, 512, 0),  Bl0, c);
      c = wmma_f16(lds_frag_a(ab, lane, 512, 32), Bl1, c);
      if (i == 0) { float bv = bL[nbase + m15];
        #pragma unroll
        for (int r = 0; r < 8; ++r) c[r] += bv; }
      #pragma unroll
      for (int j = 0; j < 8; ++j) {
        const int   k = GPK[i][j];
        const float s = GPS[i][j];
        #pragma unroll
        for (int r = 0; r < 8; ++r) gp[k][r] += s * c[r] * vrj[j][r];
      }
    }
    #pragma unroll
    for (int k = 0; k < 8; ++k)
      #pragma unroll
      for (int r = 0; r < 8; ++r)
        sA[((rt * 8 + k) * 16 + r + hi8) * 64 + nbase + m15] = (_Float16)gp[k][r];
  }
  __syncthreads();

  // ---------------- stage 2: W_gpout GEMM + norms --------------------------
  float tn[8] = {};
  for (int t = 0; t < 2; ++t) {
    const int nt = nh * 2 + t, nbase = nt * 16;
    v16h Bg0 = lds_frag_b(wg + nbase * 64, lane, 64, 0);
    v16h Bg1 = lds_frag_b(wg + nbase * 64, lane, 64, 32);
    v8f g2[8];
    #pragma unroll
    for (int k = 0; k < 8; ++k) {
      const _Float16* ab = sA + (rt * 8 + k) * 1024;    // row stride = 64 halves
      v8f c = {0.f,0.f,0.f,0.f,0.f,0.f,0.f,0.f};
      c = wmma_f16(lds_frag_a(ab, lane, 64, 0),  Bg0, c);
      c = wmma_f16(lds_frag_a(ab, lane, 64, 32), Bg1, c);
      g2[k] = c;
    }
    { float bv = bG[nbase + m15];
      #pragma unroll
      for (int r = 0; r < 8; ++r) g2[0][r] += bv; }
    #pragma unroll
    for (int r = 0; r < 8; ++r) {
      float ss = 0.f;
      #pragma unroll
      for (int k = 0; k < 8; ++k) ss += g2[k][r] * g2[k][r];
      tn[r] += sqrtf(ss);
    }
    #pragma unroll
    for (int k = 0; k < 8; ++k)
      #pragma unroll
      for (int r = 0; r < 8; ++r)
        sB[((rt * 8 + k) * 16 + r + hi8) * 64 + nbase + m15] = (_Float16)g2[k][r];
  }
  #pragma unroll
  for (int r = 0; r < 8; ++r) {
    float vsum = tn[r];
    vsum += __shfl_xor(vsum, 1, 32);
    vsum += __shfl_xor(vsum, 2, 32);
    vsum += __shfl_xor(vsum, 4, 32);
    vsum += __shfl_xor(vsum, 8, 32);
    if (m15 == 0) npart[(rt * 16 + r + hi8) * 2 + nh] = vsum;
  }
  __syncthreads();
  if (tid < 64)
    smean[tid] = (npart[tid * 2] + npart[tid * 2 + 1]) * (1.f / 64.f) + 1e-6f;
  __syncthreads();

  // ---------------- elementwise: layernorm + residual + mv-GELU ------------
  for (int idx = tid; idx < 4096; idx += 256) {
    const int row = idx >> 6, ch = idx & 63;
    const int rrt = row >> 4, r16 = row & 15;
    const float inv = 1.f / smean[row];
    const float av  = aN[ch];
    float g[8];
    #pragma unroll
    for (int k = 0; k < 8; ++k) {
      float g2v = (float)sB[((rrt * 8 + k) * 16 + r16) * 64 + ch];
      float vv0 = (float)v16[(row * 8 + k) * 64 + ch];
      g[k] = av * g2v * inv + vv0;
    }
    const float s0 = g[0];
    const float u  = 1.5957691216057308f * (s0 + 0.044715f * s0 * s0 * s0);
    const float gate = 1.f / (1.f + expf(-u));
    #pragma unroll
    for (int k = 0; k < 8; ++k)
      sA[((rrt * 8 + k) * 16 + r16) * 64 + ch] = (_Float16)(gate * g[k]);
  }
  __syncthreads();

  // ---------------- stage 3: W_wv GEMM -------------------------------------
  v8f outw[2][8];
  for (int t = 0; t < 2; ++t) {
    const int nt = nh * 2 + t, nbase = nt * 16;
    v16h Bw0 = lds_frag_b(ww + nbase * 64, lane, 64, 0);
    v16h Bw1 = lds_frag_b(ww + nbase * 64, lane, 64, 32);
    #pragma unroll
    for (int k = 0; k < 8; ++k) {
      const _Float16* ab = sA + (rt * 8 + k) * 1024;
      v8f c = {0.f,0.f,0.f,0.f,0.f,0.f,0.f,0.f};
      c = wmma_f16(lds_frag_a(ab, lane, 64, 0),  Bw0, c);
      c = wmma_f16(lds_frag_a(ab, lane, 64, 32), Bw1, c);
      outw[t][k] = c;
    }
    { float bv = bW[nbase + m15];
      #pragma unroll
      for (int r = 0; r < 8; ++r) outw[t][0][r] += bv; }
    #pragma unroll
    for (int r = 0; r < 8; ++r)
      s2vin[(rt * 16 + r + hi8) * 128 + nbase + m15] = (_Float16)outw[t][0][r];
  }
  __syncthreads();

  // ---------------- stage 4: s2v GEMM (K = 128) ----------------------------
  for (int t = 0; t < 2; ++t) {
    const int nt = nh * 2 + t, nbase = nt * 16;
    v8f sc = {0.f,0.f,0.f,0.f,0.f,0.f,0.f,0.f};
    #pragma unroll
    for (int kc = 0; kc < 4; ++kc) {
      v16h Aa = lds_frag_a(s2vin + rt * 16 * 128, lane, 128, kc * 32);
      v16h Bb = lds_frag_b(ws2v + nbase * 128, lane, 128, kc * 32);
      sc = wmma_f16(Aa, Bb, sc);
    }
    { float bv = bS[nbase + m15];
      #pragma unroll
      for (int r = 0; r < 8; ++r) sc[r] += bv; }
    outw[t][0] = sc;   // replace scalar blade
  }
  __syncthreads();     // stageA/stageB now dead -> reuse as f32 output tile

  // ---------------- output: transpose through LDS, coalesced store ---------
  float* vout = reinterpret_cast<float*>(smem);   // [64row][64ch][8k] f32 = 128KB
  for (int t = 0; t < 2; ++t) {
    const int nbase = (nh * 2 + t) * 16;
    #pragma unroll
    for (int k = 0; k < 8; ++k)
      #pragma unroll
      for (int r = 0; r < 8; ++r)
        vout[((rt * 16 + r + hi8) * 64 + nbase + m15) * 8 + k] = outw[t][k][r];
  }
  __syncthreads();
  {
    float* og = out + (size_t)Btot * 64 + (size_t)row0 * 512;
    const float4* vp = reinterpret_cast<const float4*>(vout);
    float4* op = reinterpret_cast<float4*>(og);
    for (int idx = tid; idx < 8192; idx += 256) op[idx] = vp[idx];
  }
}

extern "C" void kernel_launch(void* const* d_in, const int* in_sizes, int n_in,
                              void* d_out, int out_size, void* d_ws, size_t ws_size,
                              hipStream_t stream) {
  (void)n_in; (void)out_size; (void)d_ws; (void)ws_size;
  const float* s    = (const float*)d_in[0];
  const float* v    = (const float*)d_in[1];
  const float* Wl   = (const float*)d_in[2];
  const float* bl   = (const float*)d_in[3];
  const float* Wr   = (const float*)d_in[4];
  const float* br   = (const float*)d_in[5];
  const float* Wg   = (const float*)d_in[6];
  const float* bg   = (const float*)d_in[7];
  const float* an   = (const float*)d_in[8];
  const float* Ww   = (const float*)d_in[9];
  const float* bw   = (const float*)d_in[10];
  const float* Ws2v = (const float*)d_in[11];
  const float* bs   = (const float*)d_in[12];
  float* out = (float*)d_out;

  const int B = in_sizes[1] / 512;   // v is [B,64,8]
  (void)hipFuncSetAttribute((const void*)cvp_gp_fused,
                            hipFuncAttributeMaxDynamicSharedMemorySize,
                            SMEM_TOTAL);
  dim3 grid(B / 64), block(256);
  cvp_gp_fused<<<grid, block, SMEM_TOTAL, stream>>>(
      s, v, Wl, bl, Wr, br, Wg, bg, an, Ww, bw, Ws2v, bs, out, B);
}
